// WindowAttention_53919019434528
// MI455X (gfx1250) — compile-verified, matrix operand order checked by reading, not run on hardware
//
#include <hip/hip_runtime.h>
#include <hip/hip_bf16.h>

typedef __attribute__((ext_vector_type(16))) _Float16 v16h;
typedef __attribute__((ext_vector_type(8)))  _Float16 h8;
typedef __attribute__((ext_vector_type(4)))  _Float16 h4;
typedef __attribute__((ext_vector_type(8)))  float    v8f;

#define TPB   256
#define WAVES 8

// Padded LDS strides (fragment-row bank stride mod 64 banks is conflict-free
// while keeping 16B row alignment for b128 ds accesses).
#define SX_STRIDE  136   // halves, 272B rows  (bank stride 68 -> 4)
#define SQ_STRIDE   40   // halves,  80B rows  (bank stride 20)
#define SVT_STRIDE  72   // halves, 144B rows  (bank stride 36)
#define SS_STRIDE   68   // f32,    272B rows  (bank stride 68 -> 4)
#define SP_STRIDE  136   // halves, P stored in-place at start of each S row
#define SO_STRIDE  132   // f32, proj output reuse of sS region

// ---------------------------------------------------------------------------
// Fragment loader for v_wmma_f32_16x16x32_f16 A-operand (and B-operand given
// B^T stored row-major).  Per CDNA5 ISA 16-bit A 16x32 layout:
//   lanes 0-15 : row M=lane,    VGPR0-3 = K 0..7,  VGPR4-7 = K 16..23
//   lanes16-31 : row M=lane-16, VGPR0-3 = K 8..15, VGPR4-7 = K 24..31
// => two 16-byte loads per lane.
// ---------------------------------------------------------------------------
__device__ __forceinline__ v16h load_frag(const _Float16* __restrict__ base,
                                          int row0, int stride, int k0, int lane) {
    int half = lane >> 4;
    int r    = lane & 15;
    const _Float16* p = base + (size_t)(row0 + r) * stride + k0 + half * 8;
    h8 lo = *(const h8*)(p);
    h8 hi = *(const h8*)(p + 16);
    return __builtin_shufflevector(lo, hi, 0,1,2,3,4,5,6,7,8,9,10,11,12,13,14,15);
}

__device__ __forceinline__ v8f wmma_f16(v16h a, v16h b, v8f c) {
    return __builtin_amdgcn_wmma_f32_16x16x32_f16(false, a, false, b, (short)0, c, false, false);
}

// ---------------------------------------------------------------------------
// Prep: f32->f16 weight conversion + relative-position bias gather [4][64][64]
// ---------------------------------------------------------------------------
__global__ void prep_kernel(const float* __restrict__ qkvw,
                            const float* __restrict__ projw,
                            const float* __restrict__ btab,
                            _Float16* __restrict__ qkvw_h,
                            _Float16* __restrict__ projw_h,
                            float* __restrict__ bias) {
    int i = blockIdx.x * TPB + threadIdx.x;
    if (i < 49152) qkvw_h[i] = (_Float16)qkvw[i];
    if (i < 16384) projw_h[i] = (_Float16)projw[i];
    if (i < 16384) {
        int h = i >> 12;            // head
        int n = (i >> 6) & 63;      // query token
        int m = i & 63;             // key token
        int yi = n >> 3, xi = n & 7, yj = m >> 3, xj = m & 7;
        int rel = (yi - yj + 7) * 15 + (xi - xj + 7);
        bias[i] = btab[rel * 4 + h];
    }
}

// ---------------------------------------------------------------------------
// Main kernel: one workgroup (8 wave32) per window.  All GEMMs are computed
// transposed (D^T) so the token dimension is WMMA-N (lanes) and every D
// fragment store except the V^T one is a contiguous 16/32-byte vector store.
// ---------------------------------------------------------------------------
__global__ __launch_bounds__(TPB)
void win_attn_kernel(const float* __restrict__ x,
                     const float* __restrict__ qkv_b,
                     const float* __restrict__ proj_b,
                     const _Float16* __restrict__ qkvw,   // [384][128] f16
                     const _Float16* __restrict__ projw,  // [128][128] f16
                     const float* __restrict__ bias,      // [4][64][64] f32
                     float* __restrict__ out) {
    __shared__ __align__(16) _Float16 sX [64 * SX_STRIDE];        // 17408 B (x; later attn-out)
    __shared__ __align__(16) _Float16 sQ [4 * 64 * SQ_STRIDE];    // 20480 B [h*64+tok][d] pre-scaled
    __shared__ __align__(16) _Float16 sKk[4 * 64 * SQ_STRIDE];    // 20480 B [h*64+tok][d]
    __shared__ __align__(16) _Float16 sVT[4 * 32 * SVT_STRIDE];   // 18432 B [h*32+d][tok]
    __shared__ __align__(16) float    sS [256 * SS_STRIDE];       // 69632 B logits; P in-place; proj out

    const int tid  = threadIdx.x;
    const int lane = tid & 31;
    const int wave = __builtin_amdgcn_readfirstlane(tid >> 5);  // wave-uniform -> SGPR
    const int win  = blockIdx.x;

    // ---------------- Phase 0: load & convert x tile [64,128] --------------
    const float4* xg = (const float4*)(x + (size_t)win * 8192);
    for (int i = tid; i < 2048; i += TPB) {
        float4 t = xg[i];
        int m = i >> 5, c4 = (i & 31) * 4;
        h4 hv; hv[0] = (_Float16)t.x; hv[1] = (_Float16)t.y;
               hv[2] = (_Float16)t.z; hv[3] = (_Float16)t.w;
        *(h4*)(&sX[m * SX_STRIDE + c4]) = hv;
    }
    __syncthreads();

    // ------- Phase 1: QKV^T = Wqkv @ X^T  (96 tiles, M=chan, N=tok) --------
    const float SCALE = 0.17677669529663689f;   // 1/sqrt(32)
    for (int t = wave; t < 96; t += WAVES) {
        int mi = t / 24;                 // token tile 0..3   (N, uniform)
        int n  = t % 24;                 // channel tile 0..23 (M, uniform)
        v8f c = {};
        #pragma unroll
        for (int kk = 0; kk < 4; ++kk) {
            v16h aw = load_frag(qkvw, n  * 16, 128,       kk * 32, lane);  // A: W rows
            v16h bx = load_frag(sX,   mi * 16, SX_STRIDE, kk * 32, lane);  // B^T: X rows
            c = wmma_f16(aw, bx, c);
        }
        int tok   = mi * 16 + (lane & 15);       // N coordinate
        int rbase = (lane >> 4) * 8;             // first M (channel) of this lane
        int col0  = n * 16 + rbase;
        int which = n >> 3;                      // 0=q 1=k 2=v (uniform)
        int local = n & 7;
        int h     = local >> 1;
        int dofs  = (local & 1) * 16 + rbase;
        if (which == 0) {
            h8 st;
            #pragma unroll
            for (int r = 0; r < 8; ++r)
                st[r] = (_Float16)((c[r] + qkv_b[col0 + r]) * SCALE);
            *(h8*)&sQ[(h * 64 + tok) * SQ_STRIDE + dofs] = st;
        } else if (which == 1) {
            h8 st;
            #pragma unroll
            for (int r = 0; r < 8; ++r)
                st[r] = (_Float16)(c[r] + qkv_b[col0 + r]);
            *(h8*)&sKk[(h * 64 + tok) * SQ_STRIDE + dofs] = st;
        } else {
            #pragma unroll
            for (int r = 0; r < 8; ++r)
                sVT[(h * 32 + dofs + r) * SVT_STRIDE + tok] = (_Float16)(c[r] + qkv_b[col0 + r]);
        }
    }
    __syncthreads();

    // ------- Phase 2: S^Ttile = K @ Q^T (64 tiles, M=key, N=query) ---------
    for (int t = wave; t < 64; t += WAVES) {
        int h  = t >> 4;
        int mi = (t >> 2) & 3;           // query tile (N)
        int ni = t & 3;                  // key tile   (M)
        v16h ak = load_frag(sKk + h * 64 * SQ_STRIDE, ni * 16, SQ_STRIDE, 0, lane);
        v16h bq = load_frag(sQ  + h * 64 * SQ_STRIDE, mi * 16, SQ_STRIDE, 0, lane);
        v8f c = {};
        c = wmma_f16(ak, bq, c);
        int q     = mi * 16 + (lane & 15);
        int kbase = ni * 16 + (lane >> 4) * 8;
        float* dst = &sS[(h * 64 + q) * SS_STRIDE + kbase];  // contiguous in r
        float4 lo = {c[0], c[1], c[2], c[3]};
        float4 hi = {c[4], c[5], c[6], c[7]};
        *(float4*)(dst)     = lo;
        *(float4*)(dst + 4) = hi;
    }
    __syncthreads();

    // ---- Phase 3: softmax rows (+ bias); write P f16 in-place in own row --
    {
        int h = tid >> 6;
        int m = tid & 63;
        int row = h * 64 + m;
        const float* srow = &sS[row * SS_STRIDE];
        const float* brow = bias + row * 64;
        float v[64];
        float mx = -1e30f;
        #pragma unroll
        for (int j = 0; j < 64; ++j) { v[j] = srow[j] + brow[j]; mx = fmaxf(mx, v[j]); }
        float sum = 0.f;
        #pragma unroll
        for (int j = 0; j < 64; ++j) { float e = __expf(v[j] - mx); v[j] = e; sum += e; }
        float inv = 1.0f / sum;
        asm volatile("" ::: "memory");   // don't reorder punned stores before row reads
        _Float16* prow = (_Float16*)sS + row * SP_STRIDE;   // same row, thread-private
        #pragma unroll
        for (int j = 0; j < 64; ++j) prow[j] = (_Float16)(v[j] * inv);
    }
    __syncthreads();

    // ------- Phase 4: O^T = V^T @ P^T  (32 tiles, M=d, N=query) ------------
    const _Float16* sP = (const _Float16*)sS;
    for (int t = wave; t < 32; t += WAVES) {
        int h  = t >> 3;
        int mi = (t >> 1) & 3;           // query tile (N)
        int di = t & 1;                  // head-dim tile (M)
        v8f c = {};
        #pragma unroll
        for (int kk = 0; kk < 2; ++kk) {
            v16h av = load_frag(sVT + h * 32 * SVT_STRIDE, di * 16, SVT_STRIDE, kk * 32, lane);
            v16h bp = load_frag(sP  + h * 64 * SP_STRIDE,  mi * 16, SP_STRIDE,  kk * 32, lane);
            c = wmma_f16(av, bp, c);
        }
        int tok  = mi * 16 + (lane & 15);
        int dofs = di * 16 + (lane >> 4) * 8;
        h8 st;
        #pragma unroll
        for (int r = 0; r < 8; ++r) st[r] = (_Float16)c[r];
        *(h8*)&sX[tok * SX_STRIDE + h * 32 + dofs] = st;     // sX reused: attn-out f16
    }
    __syncthreads();

    // ------- Phase 5: out^T = Wproj @ O^T (+b) (32 tiles, M=chan, N=tok) ---
    float* sO = sS;                      // reuse logits region as f32 [64][SO_STRIDE]
    for (int t = wave; t < 32; t += WAVES) {
        int mi = t >> 3;                 // token tile (N)
        int n  = t & 7;                  // out-channel tile (M)
        v8f c = {};
        #pragma unroll
        for (int kk = 0; kk < 4; ++kk) {
            v16h aw = load_frag(projw, n  * 16, 128,       kk * 32, lane);
            v16h bo = load_frag(sX,    mi * 16, SX_STRIDE, kk * 32, lane);
            c = wmma_f16(aw, bo, c);
        }
        int tok = mi * 16 + (lane & 15);
        int cb  = n * 16 + (lane >> 4) * 8;
        float* dst = &sO[tok * SO_STRIDE + cb];              // contiguous in r
        float4 lo = {c[0] + proj_b[cb + 0], c[1] + proj_b[cb + 1],
                     c[2] + proj_b[cb + 2], c[3] + proj_b[cb + 3]};
        float4 hi = {c[4] + proj_b[cb + 4], c[5] + proj_b[cb + 5],
                     c[6] + proj_b[cb + 6], c[7] + proj_b[cb + 7]};
        *(float4*)(dst)     = lo;
        *(float4*)(dst + 4) = hi;
    }
    __syncthreads();

    // ---------------- vectorized coalesced writeout ------------------------
    float4* og4 = (float4*)(out + (size_t)win * 8192);
    for (int i = tid; i < 2048; i += TPB) {
        int m = i >> 5, c4 = (i & 31) * 4;
        og4[i] = *(const float4*)&sO[m * SO_STRIDE + c4];
    }
}

// ---------------------------------------------------------------------------
extern "C" void kernel_launch(void* const* d_in, const int* in_sizes, int n_in,
                              void* d_out, int out_size, void* d_ws, size_t ws_size,
                              hipStream_t stream) {
    const float* x      = (const float*)d_in[0];
    const float* qkv_w  = (const float*)d_in[1];
    const float* qkv_b  = (const float*)d_in[2];
    const float* proj_w = (const float*)d_in[3];
    const float* proj_b = (const float*)d_in[4];
    const float* btab   = (const float*)d_in[5];

    char* ws = (char*)d_ws;
    _Float16* qkvw_h  = (_Float16*)(ws);             // 384*128*2 = 98304 B
    _Float16* projw_h = (_Float16*)(ws + 98304);     // 128*128*2 = 32768 B
    float*    bias    = (float*)(ws + 131072);       // 4*64*64*4 = 65536 B

    prep_kernel<<<192, TPB, 0, stream>>>(qkv_w, proj_w, btab, qkvw_h, projw_h, bias);
    win_attn_kernel<<<4096, TPB, 0, stream>>>(x, qkv_b, proj_b, qkvw_h, projw_h, bias,
                                              (float*)d_out);
}
